// CrossAttention_40630390620219
// MI455X (gfx1250) — matrix operand order repaired and checked by reading, not compiled, not run on hardware
//
#include <hip/hip_runtime.h>
#include <hip/hip_bf16.h>

#define B_   8
#define T_   4096
#define D_   1024
#define H_   16
#define DH_  64
#define N_   512
#define N2_  1024
#define S_   512
#define TFD_ 256
#define AUD_ 768

typedef __attribute__((ext_vector_type(16))) __bf16 v16bf;
typedef __attribute__((ext_vector_type(8)))  __bf16 v8bf;
typedef __attribute__((ext_vector_type(8)))  float  v8f;
typedef int v4i __attribute__((vector_size(16)));

#define AS1 __attribute__((address_space(1)))
#define AS3 __attribute__((address_space(3)))

#if defined(__gfx1250__) && __has_builtin(__builtin_amdgcn_global_load_async_to_lds_b128)
#define USE_ASYNC_LDS 1
#else
#define USE_ASYNC_LDS 0
#endif

__device__ __forceinline__ float bf2f(__bf16 h) {
  unsigned short s = __builtin_bit_cast(unsigned short, h);
  unsigned u = ((unsigned)s) << 16;
  return __builtin_bit_cast(float, u);
}
__device__ __forceinline__ __bf16 f2bf(float f) {
  unsigned u = __builtin_bit_cast(unsigned, f);
  unsigned r = (u + 0x7FFFu + ((u >> 16) & 1u)) >> 16;
  unsigned short s = (unsigned short)r;
  return __builtin_bit_cast(__bf16, s);
}
__device__ __forceinline__ float loadf(float x)  { return x; }
__device__ __forceinline__ float loadf(__bf16 x) { return bf2f(x); }

// 16-byte global -> LDS copy: async DMA on gfx1250, sync fallback otherwise.
__device__ __forceinline__ void cp16(const __bf16* g, __bf16* l) {
#if USE_ASYNC_LDS
  __builtin_amdgcn_global_load_async_to_lds_b128((AS1 v4i*)g, (AS3 v4i*)l, 0, 0);
#else
  *(v8bf*)l = *(const v8bf*)g;
#endif
}
__device__ __forceinline__ void wait_async_all() {
#if USE_ASYNC_LDS
#if __has_builtin(__builtin_amdgcn_s_wait_asynccnt)
  __builtin_amdgcn_s_wait_asynccnt(0);
#else
  asm volatile("s_wait_asynccnt 0" ::: "memory");
#endif
#endif
}

// ---------------- WMMA fragment helpers (bf16 16x16x32) ----------------
// A layout (16x32 MxK): lane = lh*16 + m ; element e -> k = lh*16 + e
__device__ __forceinline__ v16bf pack_a(const __bf16* p) {   // p -> [k0 + lh*16]
  v8bf lo = *(const v8bf*)p;
  v8bf hi = *(const v8bf*)(p + 8);
  v16bf r;
#pragma unroll
  for (int i = 0; i < 8; ++i) { r[i] = lo[i]; r[i + 8] = hi[i]; }
  return r;
}
__device__ __forceinline__ v16bf load_a_frag(const __bf16* A, int lda, int row, int k0,
                                             int mi, int lh) {
  return pack_a(A + (size_t)(row + mi) * lda + (k0 + lh * 16));
}
// B layout (32x16 KxN), Bt stored [N,K] row-major: lane = lh*16 + n ; element e -> k = lh*16 + e
__device__ __forceinline__ v16bf load_b_frag(const __bf16* Bt, int ldb, int col, int k0,
                                             int ni, int lh) {
  const __bf16* p = Bt + (size_t)(col + ni) * ldb + (k0 + lh * 16);
  return *(const v16bf*)p;
}

// ---- generic GEMM: C[M,N] = A[M,K](bf16) @ Bt[N,K](bf16)^T + bias (+res) ----
// 256 threads = 8 waves (2x4), block tile 64x128, wave tile 32x32, K-step 32.
// Double-buffered LDS staging fed by async global->LDS DMA (ASYNCcnt).
template <bool OUT_BF16, bool HAS_RES>
__global__ __launch_bounds__(256) void gemm_bf16_kernel(
    const __bf16* __restrict__ A, const __bf16* __restrict__ Bt,
    const float* __restrict__ bias, const float* __restrict__ Res,
    void* __restrict__ Out, int M, int Ncols, int K) {
  __shared__ __attribute__((aligned(64))) __bf16 As[2][64 * 32];    // 8 KB
  __shared__ __attribute__((aligned(64))) __bf16 Bs[2][128 * 32];   // 16 KB
  const int tid  = threadIdx.x;
  const int w    = tid >> 5;
  const int lane = tid & 31;
  const int mi = lane & 15, lh = lane >> 4, ni = lane & 15;
  const int wm = w >> 2, wn = w & 3;
  const int rowB = blockIdx.y * 64;
  const int colB = blockIdx.x * 128;

  const int srow = tid >> 2;            // 0..63
  const int sseg = (tid & 3) * 8;       // 0,8,16,24 (bf16 elements; 16B units)

  auto stage = [&](int b, int kk) {
    // A slab: 64 rows x 32 k  (one 16B unit per thread)
    cp16(A + (size_t)(rowB + srow) * K + kk + sseg, &As[b][srow * 32 + sseg]);
    // B slab: 128 rows x 32 k (two 16B units per thread)
#pragma unroll
    for (int u = 0; u < 2; ++u) {
      const int r = srow + u * 64;
      cp16(Bt + (size_t)(colB + r) * K + kk + sseg, &Bs[b][r * 32 + sseg]);
    }
  };

  v8f c00 = {}, c01 = {}, c10 = {}, c11 = {};
  int buf = 0;
  stage(0, 0);
  for (int k0 = 0; k0 < K; k0 += 32) {
    wait_async_all();          // our async writes to As[buf]/Bs[buf] landed
    __syncthreads();           // everyone's writes visible
    if (k0 + 32 < K) stage(buf ^ 1, k0 + 32);   // overlap next slab with compute
    const __bf16* a0p = &As[buf][(wm * 32 + mi) * 32 + lh * 16];
    const __bf16* a1p = &As[buf][(wm * 32 + 16 + mi) * 32 + lh * 16];
    v16bf a0 = pack_a(a0p);
    v16bf a1 = pack_a(a1p);
    v16bf b0 = *(const v16bf*)&Bs[buf][(wn * 32 + ni) * 32 + lh * 16];
    v16bf b1 = *(const v16bf*)&Bs[buf][(wn * 32 + 16 + ni) * 32 + lh * 16];
    c00 = __builtin_amdgcn_wmma_f32_16x16x32_bf16(false, a0, false, b0, (short)0, c00, false, false);
    c01 = __builtin_amdgcn_wmma_f32_16x16x32_bf16(false, a0, false, b1, (short)0, c01, false, false);
    c10 = __builtin_amdgcn_wmma_f32_16x16x32_bf16(false, a1, false, b0, (short)0, c10, false, false);
    c11 = __builtin_amdgcn_wmma_f32_16x16x32_bf16(false, a1, false, b1, (short)0, c11, false, false);
    __syncthreads();           // done reading buf before it is overwritten
    buf ^= 1;
  }

  const int row0 = rowB + wm * 32;
  const int col0 = colB + wn * 32;
  const v8f cc[4] = {c00, c01, c10, c11};
#pragma unroll
  for (int t = 0; t < 4; ++t) {
    const int trow = row0 + (t >> 1) * 16;
    const int tcol = col0 + (t & 1) * 16;
    const int col  = tcol + ni;
    const float bv = bias[col];
#pragma unroll
    for (int r = 0; r < 8; ++r) {
      const int row = trow + lh * 8 + r;     // C layout: vgpr r -> m = r + 8*(lane/16)
      const size_t idx = (size_t)row * Ncols + col;
      float v = cc[t][r] + bv;
      if (HAS_RES) v += Res[idx];
      if (OUT_BF16) ((__bf16*)Out)[idx] = f2bf(v);
      else          ((float*)Out)[idx]  = v;
    }
  }
}

// ------- batched per-(b,h) GEMM: ys[b,t,h,:] = silu(q[b,t,h,:] @ attnT[b,h]^T + ssum[b,h,:]) ----
// grid = (T/128, B*H), wave handles 16 rows x 64 cols, K = 64. attnT is tiny -> L2 resident.
__global__ __launch_bounds__(256) void gemm_qattn_kernel(
    const __bf16* __restrict__ q, const __bf16* __restrict__ attnT,
    const float* __restrict__ ssum, __bf16* __restrict__ ys) {
  const int bh = blockIdx.y;
  const int b = bh >> 4, h = bh & 15;
  const int tid = threadIdx.x;
  const int w = tid >> 5, lane = tid & 31;
  const int mi = lane & 15, lh = lane >> 4, ni = lane & 15;
  const int row0 = blockIdx.x * 128 + w * 16;
  const __bf16* Ab = q + (size_t)b * T_ * D_ + h * DH_;      // lda = D
  const __bf16* Bt = attnT + (size_t)bh * DH_ * DH_;         // [l, d], ldb = 64
  const float*  sb = ssum + (size_t)bh * DH_;
  v8f c0 = {}, c1 = {}, c2 = {}, c3 = {};
#pragma unroll
  for (int k0 = 0; k0 < DH_; k0 += 32) {
    v16bf a  = load_a_frag(Ab, D_, row0, k0, mi, lh);
    v16bf b0 = load_b_frag(Bt, DH_, 0,  k0, ni, lh);
    v16bf b1 = load_b_frag(Bt, DH_, 16, k0, ni, lh);
    v16bf b2 = load_b_frag(Bt, DH_, 32, k0, ni, lh);
    v16bf b3 = load_b_frag(Bt, DH_, 48, k0, ni, lh);
    c0 = __builtin_amdgcn_wmma_f32_16x16x32_bf16(false, a, false, b0, (short)0, c0, false, false);
    c1 = __builtin_amdgcn_wmma_f32_16x16x32_bf16(false, a, false, b1, (short)0, c1, false, false);
    c2 = __builtin_amdgcn_wmma_f32_16x16x32_bf16(false, a, false, b2, (short)0, c2, false, false);
    c3 = __builtin_amdgcn_wmma_f32_16x16x32_bf16(false, a, false, b3, (short)0, c3, false, false);
  }
  __bf16* Ob = ys + (size_t)b * T_ * D_ + h * DH_;           // ldc = D
  const v8f cc[4] = {c0, c1, c2, c3};
#pragma unroll
  for (int t = 0; t < 4; ++t) {
    const int col = t * 16 + ni;
    const float bv = sb[col];
#pragma unroll
    for (int r = 0; r < 8; ++r) {
      const int row = row0 + lh * 8 + r;
      float v = cc[t][r] + bv;
      v = v / (1.f + expf(-v));                              // SiLU
      Ob[(size_t)row * D_ + col] = f2bf(v);
    }
  }
}

// ---------------- LayerNorm over D=1024, one block per row, out bf16 ----------------
template <typename TIN>
__global__ __launch_bounds__(256) void ln_kernel(const TIN* __restrict__ X,
                                                 const float* __restrict__ g,
                                                 const float* __restrict__ bta,
                                                 __bf16* __restrict__ Out) {
  const int row = blockIdx.x;
  const int tid = threadIdx.x;
  __shared__ float red[256];
  float v[4];
  float s = 0.f;
#pragma unroll
  for (int i = 0; i < 4; ++i) {
    float x = loadf(X[(size_t)row * D_ + tid + i * 256]);
    v[i] = x; s += x;
  }
  red[tid] = s; __syncthreads();
  for (int o = 128; o > 0; o >>= 1) { if (tid < o) red[tid] += red[tid + o]; __syncthreads(); }
  const float mean = red[0] / (float)D_;
  __syncthreads();
  float s2 = 0.f;
#pragma unroll
  for (int i = 0; i < 4; ++i) { float d = v[i] - mean; s2 += d * d; }
  red[tid] = s2; __syncthreads();
  for (int o = 128; o > 0; o >>= 1) { if (tid < o) red[tid] += red[tid + o]; __syncthreads(); }
  const float rstd = rsqrtf(red[0] / (float)D_ + 1e-5f);
#pragma unroll
  for (int i = 0; i < 4; ++i) {
    const int c = tid + i * 256;
    Out[(size_t)row * D_ + c] = f2bf((v[i] - mean) * rstd * g[c] + bta[c]);
  }
}

// ------ softmax over feature dim (64 contiguous) in place: one thread per (b,t,h) ------
__global__ void softmax_feat_kernel(__bf16* __restrict__ q, int total) {
  const int i = blockIdx.x * blockDim.x + threadIdx.x;
  if (i >= total) return;
  __bf16* p = q + (size_t)i * DH_;
  float m = -3.4e38f;
  for (int j = 0; j < DH_; ++j) m = fmaxf(m, bf2f(p[j]));
  float s = 0.f;
  for (int j = 0; j < DH_; ++j) s += expf(bf2f(p[j]) - m);
  const float inv = 1.f / s;
  for (int j = 0; j < DH_; ++j) p[j] = f2bf(expf(bf2f(p[j]) - m) * inv);
}

// ------ softmax over sequence dim in place: one thread per (b, channel) column ------
__global__ void softmax_seq_kernel(__bf16* __restrict__ kb, int rows, int total) {
  const int i = blockIdx.x * blockDim.x + threadIdx.x;
  if (i >= total) return;                       // total = B * D
  const int b = i / D_, c = i % D_;
  __bf16* base = kb + (size_t)b * rows * D_ + c;
  float m = -3.4e38f;
  for (int n = 0; n < rows; ++n) m = fmaxf(m, bf2f(base[(size_t)n * D_]));
  float s = 0.f;
  for (int n = 0; n < rows; ++n) s += expf(bf2f(base[(size_t)n * D_]) - m);
  const float inv = 1.f / s;
  for (int n = 0; n < rows; ++n)
    base[(size_t)n * D_] = f2bf(expf(bf2f(base[(size_t)n * D_]) - m) * inv);
}

// ------ sksum[b,s,h] = sum_d sk[b,s,h,d] ------
__global__ void sksum_kernel(const __bf16* __restrict__ sk, float* __restrict__ sksum, int total) {
  const int i = blockIdx.x * blockDim.x + threadIdx.x;
  if (i >= total) return;                       // total = B*S*H
  const __bf16* p = sk + (size_t)i * DH_;
  float s = 0.f;
  for (int j = 0; j < DH_; ++j) s += bf2f(p[j]);
  sksum[i] = s;
}

// ------ ssum[b,h,l] = sum_s sksum[b,s,h] * sv[b,s,h,l]  (== sum_s sattn) ------
__global__ void ssum_kernel(const float* __restrict__ sksum, const __bf16* __restrict__ sv,
                            float* __restrict__ ssum) {
  const int i = blockIdx.x * blockDim.x + threadIdx.x;
  if (i >= B_ * H_ * DH_) return;
  const int l = i & 63, h = (i >> 6) & 15, b = i >> 10;
  float acc = 0.f;
  for (int s = 0; s < S_; ++s)
    acc += sksum[((size_t)b * S_ + s) * H_ + h] *
           bf2f(sv[((size_t)b * S_ + s) * D_ + h * DH_ + l]);
  ssum[i] = acc;
}

// ------ attnT[b,h,l,d] = sum_n k[b,n,h,d] * v[b,n,h,l]  (store transposed for GEMM B) ------
__global__ __launch_bounds__(256) void attn_kernel(const __bf16* __restrict__ kb,
                                                   const __bf16* __restrict__ vb,
                                                   __bf16* __restrict__ attnT) {
  const int bh = blockIdx.x, b = bh >> 4, h = bh & 15;
  const int tid = threadIdx.x;
  __shared__ float ks[8][64], vs[8][64];
  const int l = tid >> 2, d0 = (tid & 3) * 16;
  float acc[16] = {};
  const int lr = tid >> 5, lc = (tid & 31) * 2;
  for (int n0 = 0; n0 < N2_; n0 += 8) {
    __syncthreads();
    const size_t off = ((size_t)b * N2_ + n0 + lr) * D_ + h * DH_ + lc;
    ks[lr][lc] = bf2f(kb[off]); ks[lr][lc + 1] = bf2f(kb[off + 1]);
    vs[lr][lc] = bf2f(vb[off]); vs[lr][lc + 1] = bf2f(vb[off + 1]);
    __syncthreads();
#pragma unroll
    for (int n = 0; n < 8; ++n) {
      const float vv = vs[n][l];
#pragma unroll
      for (int j = 0; j < 16; ++j) acc[j] += ks[n][d0 + j] * vv;
    }
  }
  __bf16* out = attnT + ((size_t)bh * DH_ + l) * DH_ + d0;
#pragma unroll
  for (int j = 0; j < 16; ++j) out[j] = f2bf(acc[j]);
}

// ------ weight transpose + bf16 convert: W[K,N] f32 -> Wt[N,K] bf16 ------
__global__ void wt_kernel(const float* __restrict__ W, __bf16* __restrict__ Wt, int K, int Nc) {
  const int i = blockIdx.x * blockDim.x + threadIdx.x;
  if (i >= K * Nc) return;
  const int k = i / Nc, n = i % Nc;
  Wt[(size_t)n * K + k] = f2bf(W[i]);
}

// ------ xf_proj = xf @ Wa + ba   ([8,768]x[768,256], tiny) ------
__global__ void audio_proj_kernel(const float* __restrict__ xf, const float* __restrict__ Wa,
                                  const float* __restrict__ ba, float* __restrict__ xfp) {
  const int i = blockIdx.x * blockDim.x + threadIdx.x;
  if (i >= B_ * TFD_) return;
  const int b = i / TFD_, c = i % TFD_;
  float acc = ba[c];
  for (int k = 0; k < AUD_; ++k) acc += xf[(size_t)b * AUD_ + k] * Wa[(size_t)k * TFD_ + c];
  xfp[i] = acc;
}

// ------ Cin[b, r, c] = r < N ? xw[b,r,c] : xf_proj[b,c]  (bf16) ------
__global__ void cin_kernel(const float* __restrict__ xw, const float* __restrict__ xfp,
                           __bf16* __restrict__ Cin, int total) {
  const int i = blockIdx.x * blockDim.x + threadIdx.x;
  if (i >= total) return;
  const int c = i % TFD_;
  const int r = (i / TFD_) % (2 * N_);
  const int b = i / (TFD_ * 2 * N_);
  const float v = (r < N_) ? xw[((size_t)b * N_ + r) * TFD_ + c] : xfp[(size_t)b * TFD_ + c];
  Cin[i] = f2bf(v);
}

extern "C" void kernel_launch(void* const* d_in, const int* in_sizes, int n_in,
                              void* d_out, int out_size, void* d_ws, size_t ws_size,
                              hipStream_t stream) {
  (void)in_sizes; (void)n_in; (void)out_size; (void)ws_size;
  const float* x       = (const float*)d_in[0];
  const float* xf      = (const float*)d_in[1];
  const float* xw      = (const float*)d_in[2];
  const float* xs      = (const float*)d_in[3];
  const float* norm_g  = (const float*)d_in[4];
  const float* norm_b  = (const float*)d_in[5];
  const float* tnorm_g = (const float*)d_in[6];
  const float* tnorm_b = (const float*)d_in[7];
  const float* snorm_g = (const float*)d_in[8];
  const float* snorm_b = (const float*)d_in[9];
  const float* Wq = (const float*)d_in[10]; const float* bq = (const float*)d_in[11];
  const float* Wk = (const float*)d_in[12]; const float* bk = (const float*)d_in[13];
  const float* Wv = (const float*)d_in[14]; const float* bv = (const float*)d_in[15];
  const float* Wa = (const float*)d_in[16]; const float* ba = (const float*)d_in[17];
  const float* Wat = (const float*)d_in[18]; const float* bat = (const float*)d_in[19];
  const float* Wo = (const float*)d_in[20]; const float* bo = (const float*)d_in[21];

  char* base = (char*)d_ws;
  size_t off = 0;
  auto take = [&](size_t bytes) -> void* {
    void* r = base + off;
    off = (off + bytes + 255) & ~(size_t)255;
    return r;
  };

  __bf16* WqT  = (__bf16*)take((size_t)D_ * D_ * 2);
  __bf16* WkT  = (__bf16*)take((size_t)D_ * D_ * 2);
  __bf16* WvT  = (__bf16*)take((size_t)D_ * D_ * 2);
  __bf16* WoT  = (__bf16*)take((size_t)D_ * D_ * 2);
  __bf16* WatT = (__bf16*)take((size_t)TFD_ * D_ * 2);
  float*  xfp  = (float*) take((size_t)B_ * TFD_ * 4);
  __bf16* Cin  = (__bf16*)take((size_t)B_ * 2 * N_ * TFD_ * 2);
  __bf16* tn   = (__bf16*)take((size_t)B_ * N2_ * D_ * 2);
  __bf16* sn   = (__bf16*)take((size_t)B_ * S_ * D_ * 2);
  __bf16* xn   = (__bf16*)take((size_t)B_ * T_ * D_ * 2);
  __bf16* kbuf = (__bf16*)take((size_t)B_ * N2_ * D_ * 2);
  __bf16* vbuf = (__bf16*)take((size_t)B_ * N2_ * D_ * 2);
  __bf16* skb  = (__bf16*)take((size_t)B_ * S_ * D_ * 2);
  __bf16* svb  = (__bf16*)take((size_t)B_ * S_ * D_ * 2);
  float*  sks  = (float*) take((size_t)B_ * S_ * H_ * 4);
  __bf16* attnT= (__bf16*)take((size_t)B_ * H_ * DH_ * DH_ * 2);
  float*  ssum = (float*) take((size_t)B_ * H_ * DH_ * 4);
  __bf16* qbuf = (__bf16*)take((size_t)B_ * T_ * D_ * 2);
  __bf16* ys   = (__bf16*)take((size_t)B_ * T_ * D_ * 2);

  const int TB = 256;

  // 1) weight transposes -> bf16 [N,K]
  wt_kernel<<<(D_ * D_ + TB - 1) / TB, TB, 0, stream>>>(Wq, WqT, D_, D_);
  wt_kernel<<<(D_ * D_ + TB - 1) / TB, TB, 0, stream>>>(Wk, WkT, D_, D_);
  wt_kernel<<<(D_ * D_ + TB - 1) / TB, TB, 0, stream>>>(Wv, WvT, D_, D_);
  wt_kernel<<<(D_ * D_ + TB - 1) / TB, TB, 0, stream>>>(Wo, WoT, D_, D_);
  wt_kernel<<<(TFD_ * D_ + TB - 1) / TB, TB, 0, stream>>>(Wat, WatT, TFD_, D_);

  // 2) audio projection + concat input
  audio_proj_kernel<<<(B_ * TFD_ + TB - 1) / TB, TB, 0, stream>>>(xf, Wa, ba, xfp);
  {
    const int tot = B_ * 2 * N_ * TFD_;
    cin_kernel<<<(tot + TB - 1) / TB, TB, 0, stream>>>(xw, xfp, Cin, tot);
  }

  // 3) tn_raw = Cin @ WatT + bat   (M=8192, N=1024, K=256) -> bf16, then LN in place
  gemm_bf16_kernel<true, false><<<dim3(D_ / 128, (B_ * N2_) / 64), TB, 0, stream>>>(
      Cin, WatT, bat, nullptr, tn, B_ * N2_, D_, TFD_);
  ln_kernel<__bf16><<<B_ * N2_, TB, 0, stream>>>(tn, tnorm_g, tnorm_b, tn);

  // 4) sn = LN(xs), xn = LN(x)
  ln_kernel<float><<<B_ * S_, TB, 0, stream>>>(xs, snorm_g, snorm_b, sn);
  ln_kernel<float><<<B_ * T_, TB, 0, stream>>>(x, norm_g, norm_b, xn);

  // 5) k/v/sk/sv projections (all bf16 out)
  gemm_bf16_kernel<true, false><<<dim3(D_ / 128, (B_ * N2_) / 64), TB, 0, stream>>>(
      tn, WkT, bk, nullptr, kbuf, B_ * N2_, D_, D_);
  gemm_bf16_kernel<true, false><<<dim3(D_ / 128, (B_ * N2_) / 64), TB, 0, stream>>>(
      tn, WvT, bv, nullptr, vbuf, B_ * N2_, D_, D_);
  gemm_bf16_kernel<true, false><<<dim3(D_ / 128, (B_ * S_) / 64), TB, 0, stream>>>(
      sn, WkT, bk, nullptr, skb, B_ * S_, D_, D_);
  gemm_bf16_kernel<true, false><<<dim3(D_ / 128, (B_ * S_) / 64), TB, 0, stream>>>(
      sn, WvT, bv, nullptr, svb, B_ * S_, D_, D_);

  // 6) sequence softmax in place (k over 2N rows, sk over S rows)
  softmax_seq_kernel<<<(B_ * D_ + TB - 1) / TB, TB, 0, stream>>>(kbuf, N2_, B_ * D_);
  softmax_seq_kernel<<<(B_ * D_ + TB - 1) / TB, TB, 0, stream>>>(skb, S_, B_ * D_);

  // 7) collapse sattn: sksum then ssum
  sksum_kernel<<<(B_ * S_ * H_ + TB - 1) / TB, TB, 0, stream>>>(skb, sks, B_ * S_ * H_);
  ssum_kernel<<<(B_ * H_ * DH_ + TB - 1) / TB, TB, 0, stream>>>(sks, svb, ssum);

  // 8) attnT[b,h,l,d] = sum_n k*v
  attn_kernel<<<B_ * H_, TB, 0, stream>>>(kbuf, vbuf, attnT);

  // 9) q = softmax_feat(xn @ WqT + bq)
  gemm_bf16_kernel<true, false><<<dim3(D_ / 128, (B_ * T_) / 64), TB, 0, stream>>>(
      xn, WqT, bq, nullptr, qbuf, B_ * T_, D_, D_);
  softmax_feat_kernel<<<(B_ * T_ * H_ + TB - 1) / TB, TB, 0, stream>>>(qbuf, B_ * T_ * H_);

  // 10) ys = silu(q @ attnT^T + ssum)   (batched per (b,h))
  gemm_qattn_kernel<<<dim3(T_ / 128, B_ * H_), TB, 0, stream>>>(qbuf, attnT, ssum, ys);

  // 11) out = x + ys @ WoT + bo  (f32 out, residual)
  gemm_bf16_kernel<false, true><<<dim3(D_ / 128, (B_ * T_) / 64), TB, 0, stream>>>(
      ys, WoT, bo, x, (float*)d_out, B_ * T_, D_, D_);
}
